// RelativeMultiHeadAttention_36069135351842
// MI455X (gfx1250) — compile-verified, matrix operand order checked by reading, not run on hardware
//
#include <hip/hip_runtime.h>
#include <math.h>

// ---------------------------------------------------------------------------
// Relative Multi-Head Attention (Transformer-XL style) for MI455X / gfx1250.
// wave32, WMMA bf16 16x16x32 path. B=2, S=2048, D=512, H=16, Dh=32.
// ---------------------------------------------------------------------------

typedef __bf16 bf16;
typedef __attribute__((ext_vector_type(16))) __bf16 v16bf;
typedef __attribute__((ext_vector_type(8)))  float  v8f;
typedef __attribute__((ext_vector_type(4)))  float  v4f;
typedef __attribute__((ext_vector_type(4)))  int    v4i;

#define S_LEN 2048
#define D_MODEL 512
#define N_HEADS 16
#define D_HEAD 32
#define BS 2
#define M_TOTAL (BS * S_LEN)            // 4096 rows for projections
static __device__ __constant__ float kInvSqrtD = 0.04419417382415922f; // 1/sqrt(512)

__device__ __forceinline__ int lane_id() { return threadIdx.x & 31; }
__device__ __forceinline__ int wave_id() { return threadIdx.x >> 5; }

// 16-bit A/B fragment K index for element e (0..15) of the per-lane v16bf,
// per CDNA5 ISA 7.12.2: e<8 -> K = lanehalf*8 + e ; e>=8 -> K = 16 + lanehalf*8 + (e-8)
__device__ __forceinline__ int kidx16(int lane, int e) {
  int v = e >> 1, b = e & 1;
  return ((v & 4) << 2) | (((lane >> 4) & 1) << 3) | ((v & 3) << 1) | b;
}

// 32-bit 16x16 C/D layout: VGPR v of lane L holds (M = v + 8*(L>=16), N = L%16)
__device__ __forceinline__ void cd_coord(int lane, int v, int& m, int& n) {
  m = v + ((lane >> 4) << 3);
  n = lane & 15;
}

__device__ __forceinline__ v8f wmma_bf16(v16bf a, v16bf b, v8f c) {
  return __builtin_amdgcn_wmma_f32_16x16x32_bf16(
      /*neg_a=*/false, a, /*neg_b=*/false, b,
      /*c_mod=*/(short)0, c, /*reuse_a=*/false, /*reuse_b=*/false);
}

// Because kidx16 is contiguous within each 8-element half, a fragment whose
// source row is contiguous in memory is exactly two 16-byte loads per lane.
union FragBits { struct { v4i lo, hi; } p; v16bf f; };

__device__ __forceinline__ v16bf frag_from_bf16row(const bf16* __restrict__ rowp,
                                                   int lanehalf) {
  FragBits u;
  u.p.lo = *(const v4i*)(rowp + (lanehalf << 3));        // K = lh*8 + 0..7
  u.p.hi = *(const v4i*)(rowp + 16 + (lanehalf << 3));   // K = 16 + lh*8 + 0..7
  return u.f;
}

__device__ __forceinline__ v16bf frag_from_f32row(const float* __restrict__ rowp,
                                                  int lanehalf) {
  const v4f a0 = *(const v4f*)(rowp + (lanehalf << 3));
  const v4f a1 = *(const v4f*)(rowp + (lanehalf << 3) + 4);
  const v4f b0 = *(const v4f*)(rowp + 16 + (lanehalf << 3));
  const v4f b1 = *(const v4f*)(rowp + 16 + (lanehalf << 3) + 4);
  v16bf f;
  f[0]  = (bf16)a0.x; f[1]  = (bf16)a0.y; f[2]  = (bf16)a0.z; f[3]  = (bf16)a0.w;
  f[4]  = (bf16)a1.x; f[5]  = (bf16)a1.y; f[6]  = (bf16)a1.z; f[7]  = (bf16)a1.w;
  f[8]  = (bf16)b0.x; f[9]  = (bf16)b0.y; f[10] = (bf16)b0.z; f[11] = (bf16)b0.w;
  f[12] = (bf16)b1.x; f[13] = (bf16)b1.y; f[14] = (bf16)b1.z; f[15] = (bf16)b1.w;
  return f;
}

// ---------------------------------------------------------------------------
// Kernel 1: the four input projections. out[s,e] = sum_d X[s,d]*W[e,d] (+bias)
//   z=0: q  -> store qu = bf16(q + u_bias) and qv = bf16(q + v_bias)
//   z=1: k  -> kb     z=2: v -> vb     z=3: pos (no bias) -> peb
// ---------------------------------------------------------------------------
__global__ void proj_kernel(const float* __restrict__ query,
                            const float* __restrict__ key,
                            const float* __restrict__ value,
                            const float* __restrict__ pos,
                            const float* __restrict__ Wq,
                            const float* __restrict__ Wk,
                            const float* __restrict__ Wv,
                            const float* __restrict__ Wpos,
                            const float* __restrict__ bq,
                            const float* __restrict__ bk,
                            const float* __restrict__ bv,
                            const float* __restrict__ u_bias,
                            const float* __restrict__ v_bias,
                            bf16* __restrict__ qu, bf16* __restrict__ qv,
                            bf16* __restrict__ kb, bf16* __restrict__ vb,
                            bf16* __restrict__ peb) {
  const int lane = lane_id();
  const int wave = wave_id();
  const int z    = blockIdx.z;
  const int row0 = blockIdx.y * 16;
  const int col0 = (blockIdx.x * 8 + wave) * 16;
  const int mrow = lane & 15;
  const int lh   = lane >> 4;

  const float* X    = (z == 0) ? query : (z == 1) ? key : (z == 2) ? value : pos;
  const float* W    = (z == 0) ? Wq    : (z == 1) ? Wk  : (z == 2) ? Wv    : Wpos;
  const float* bias = (z == 0) ? bq    : (z == 1) ? bk  : (z == 2) ? bv    : nullptr;

  const float* Xrow = X + (size_t)(row0 + mrow) * D_MODEL;
  const float* Wrow = W + (size_t)(col0 + mrow) * D_MODEL;   // B col n = W row n

  v8f acc = {};
  for (int k0 = 0; k0 < D_MODEL; k0 += 32) {
    v16bf a  = frag_from_f32row(Xrow + k0, lh);
    v16bf bw = frag_from_f32row(Wrow + k0, lh);
    acc = wmma_bf16(a, bw, acc);
  }
#pragma unroll
  for (int v = 0; v < 8; ++v) {
    int mm, nn; cd_coord(lane, v, mm, nn);
    const int row = row0 + mm, col = col0 + nn;
    const size_t idx = (size_t)row * D_MODEL + col;
    float val = acc[v] + (bias ? bias[col] : 0.0f);
    if (z == 0) {
      qu[idx] = (bf16)(val + u_bias[col]);  // u_bias flat [H*Dh] == [e]
      qv[idx] = (bf16)(val + v_bias[col]);
    } else if (z == 1) kb[idx]  = (bf16)val;
    else if (z == 2)   vb[idx]  = (bf16)val;
    else               peb[idx] = (bf16)val;
  }
}

// ---------------------------------------------------------------------------
// Kernel 2: fused score + relative shift + mask + softmax.
// One block per (bh, i-block of 16 rows). LDS holds:
//   ps_lds[17][2048]  pos scores for rows i0..i0+16 (band needed by shift)
//   sc_lds[16][2048]  pre-softmax scores, softmaxed in place, then streamed out
// Head K-dim is exactly 32 -> a single WMMA per 16x16 tile.
// ---------------------------------------------------------------------------
__global__ void score_softmax_kernel(const bf16* __restrict__ qu,
                                     const bf16* __restrict__ qv,
                                     const bf16* __restrict__ kb,
                                     const bf16* __restrict__ peb,
                                     const unsigned char* __restrict__ mask,
                                     float* __restrict__ attn) {
  extern __shared__ float smem[];
  float* ps_lds = smem;                  // 17 * 2048 floats
  float* sc_lds = smem + 17 * S_LEN;     // 16 * 2048 floats

  const int lane = lane_id();
  const int wave = wave_id();
  const int i0   = blockIdx.x * 16;
  const int bh   = blockIdx.y;
  const int b    = bh >> 4, h = bh & 15;
  const int mrow = lane & 15;
  const int lh   = lane >> 4;

  const size_t base = (size_t)b * S_LEN * D_MODEL + (size_t)h * D_HEAD;
  const bf16* quR = qu  + base;
  const bf16* qvR = qv  + base;
  const bf16* kR  = kb  + base;
  const bf16* peR = peb + base;

  // A fragments (rows i0..i0+15, K = 32 = full head dim), loaded once.
  const v16bf aqv = frag_from_bf16row(qvR + (size_t)(i0 + mrow) * D_MODEL, lh);
  const v16bf aqu = frag_from_bf16row(quR + (size_t)(i0 + mrow) * D_MODEL, lh);

  // ---- Phase 1: pos scores ps[i0..i0+15][*] via WMMA -----------------------
  for (int jt = wave; jt < S_LEN / 16; jt += 8) {
    const int j0 = jt * 16;
    v16bf bpe = frag_from_bf16row(peR + (size_t)(j0 + mrow) * D_MODEL, lh);
    v8f acc = {};
    acc = wmma_bf16(aqv, bpe, acc);
#pragma unroll
    for (int v = 0; v < 8; ++v) {
      int mm, nn; cd_coord(lane, v, mm, nn);
      ps_lds[mm * S_LEN + (j0 + nn)] = acc[v];
    }
  }
  // ---- Phase 1b: ps row 16 (global row i0+16): tiny scalar dots -----------
  {
    const int irow = i0 + 16;
    for (int j = threadIdx.x; j < S_LEN; j += 256) {
      float s = 0.0f;
      if (irow < S_LEN) {
#pragma unroll
        for (int d = 0; d < D_HEAD; ++d)
          s += (float)qvR[(size_t)irow * D_MODEL + d] *
               (float)peR[(size_t)j    * D_MODEL + d];
      }
      ps_lds[16 * S_LEN + j] = s;
    }
  }
  __syncthreads();

  // ---- Phase 2: content WMMA + rel-shift + mask + scale -> sc_lds ---------
  for (int jt = wave; jt < S_LEN / 16; jt += 8) {
    const int j0 = jt * 16;
    v16bf bkf = frag_from_bf16row(kR + (size_t)(j0 + mrow) * D_MODEL, lh);
    v8f acc = {};
    acc = wmma_bf16(aqu, bkf, acc);
#pragma unroll
    for (int v = 0; v < 8; ++v) {
      int mm, nn; cd_coord(lane, v, mm, nn);
      const int i = i0 + mm, j = j0 + nn;
      // _rel_shift: pad zero col, reshape (S+1,S), drop row 0, reshape (S,S).
      float sp;
      if (j <= i)            sp = ps_lds[mm * S_LEN + (S_LEN - 1 - i + j)];
      else if (j == i + 1)   sp = 0.0f;
      else                   sp = ps_lds[(mm + 1) * S_LEN + (j - i - 2)];
      float s = (acc[v] + sp) * kInvSqrtD;
      if (mask[(size_t)b * S_LEN + j]) s = -1e9f;
      sc_lds[mm * S_LEN + j] = s;
    }
  }
  __syncthreads();

  // ---- Phase 3: per-row softmax (wave-per-row, shuffle reductions) --------
  for (int r = wave; r < 16; r += 8) {
    float* row = sc_lds + r * S_LEN;
    float mx = -3.4e38f;
    for (int j = lane; j < S_LEN; j += 32) mx = fmaxf(mx, row[j]);
#pragma unroll
    for (int off = 16; off > 0; off >>= 1) mx = fmaxf(mx, __shfl_xor(mx, off, 32));
    float sum = 0.0f;
    for (int j = lane; j < S_LEN; j += 32) {
      const float e = __expf(row[j] - mx);
      row[j] = e;
      sum += e;
    }
#pragma unroll
    for (int off = 16; off > 0; off >>= 1) sum += __shfl_xor(sum, off, 32);
    const float inv = 1.0f / sum;
    float* dst = attn + ((size_t)bh * S_LEN + (i0 + r)) * S_LEN;
    for (int j = lane; j < S_LEN; j += 32) dst[j] = row[j] * inv;
  }
}

// ---------------------------------------------------------------------------
// Kernel 3: context = attn @ v   -> ctx[b, i, h*32 + d]  (f32)
// Block = 8 waves x 16 rows = 128 rows of one (b,h); each wave does both
// 16-wide n-tiles of the head. Per K-step the shared 32x32 bf16 V panel is
// staged into LDS with CDNA5 async-to-LDS copies (ASYNCcnt) and the
// transposed B fragments are built from LDS.
// ---------------------------------------------------------------------------
__global__ void context_kernel(const float* __restrict__ attn,
                               const bf16* __restrict__ vb,
                               float* __restrict__ ctx) {
  __shared__ bf16 vpanel[32 * 32];      // [k][n], 2 KB, LDS alloc offset 0

  const int lane = lane_id();
  const int wave = wave_id();
  const int bh   = blockIdx.y;
  const int b    = bh >> 4, h = bh & 15;
  const int row0 = (blockIdx.x * 8 + wave) * 16;
  const int mrow = lane & 15;
  const int lh   = lane >> 4;

  const float* Arow = attn + (size_t)bh * S_LEN * S_LEN
                           + (size_t)(row0 + mrow) * S_LEN;
  const bf16*  V0   = vb + (size_t)b * S_LEN * D_MODEL + h * D_HEAD;

  v8f acc0 = {}, acc1 = {};
  for (int k0 = 0; k0 < S_LEN; k0 += 32) {
    // ---- async-stage V[k0..k0+31][0..31] into LDS (128 x b128 copies) ----
    if (threadIdx.x < 128) {
      const int r = threadIdx.x >> 2;            // k row 0..31
      const int c = (threadIdx.x & 3) << 3;      // 8-elem chunk 0..3
      const bf16* g = V0 + (size_t)(k0 + r) * D_MODEL + c;
      const unsigned loff = (unsigned)((r * 32 + c) * sizeof(bf16));
      asm volatile("global_load_async_to_lds_b128 %0, %1, off"
                   :: "v"(loff), "v"(g) : "memory");
    }
    asm volatile("s_wait_asynccnt 0x0" ::: "memory");
    __syncthreads();

    if (k0 + 128 < S_LEN) __builtin_prefetch(Arow + k0 + 128, 0, 1);

    // A fragment from f32 attn row (two contiguous 32B spans + cvt).
    v16bf a = frag_from_f32row(Arow + k0, lh);

    // B fragments gathered from the LDS panel (transposed access).
    v16bf b0, b1;
#pragma unroll
    for (int e = 0; e < 16; ++e) {
      const int kk = kidx16(lane, e);
      b0[e] = vpanel[kk * 32 + mrow];
      b1[e] = vpanel[kk * 32 + 16 + mrow];
    }
    acc0 = wmma_bf16(a, b0, acc0);
    acc1 = wmma_bf16(a, b1, acc1);
    __syncthreads();   // vpanel is overwritten next iteration
  }
#pragma unroll
  for (int v = 0; v < 8; ++v) {
    int mm, nn; cd_coord(lane, v, mm, nn);
    const size_t rowi = (size_t)(b * S_LEN + row0 + mm) * D_MODEL + h * D_HEAD;
    ctx[rowi + nn]      = acc0[v];
    ctx[rowi + 16 + nn] = acc1[v];
  }
}

// ---------------------------------------------------------------------------
// Kernel 4: out = ctx @ Wo^T + bo   (f32 result into d_out)
// ---------------------------------------------------------------------------
__global__ void outproj_kernel(const float* __restrict__ ctx,
                               const float* __restrict__ Wo,
                               const float* __restrict__ bo,
                               float* __restrict__ out) {
  const int lane = lane_id();
  const int wave = wave_id();
  const int row0 = blockIdx.y * 16;
  const int col0 = (blockIdx.x * 8 + wave) * 16;
  const int mrow = lane & 15;
  const int lh   = lane >> 4;

  const float* Crow = ctx + (size_t)(row0 + mrow) * D_MODEL;
  const float* Wrow = Wo + (size_t)(col0 + mrow) * D_MODEL;

  v8f acc = {};
  for (int k0 = 0; k0 < D_MODEL; k0 += 32) {
    v16bf a  = frag_from_f32row(Crow + k0, lh);
    v16bf bw = frag_from_f32row(Wrow + k0, lh);
    acc = wmma_bf16(a, bw, acc);
  }
#pragma unroll
  for (int v = 0; v < 8; ++v) {
    int mm, nn; cd_coord(lane, v, mm, nn);
    out[(size_t)(row0 + mm) * D_MODEL + col0 + nn] = acc[v] + bo[col0 + nn];
  }
}

// ---------------------------------------------------------------------------
extern "C" void kernel_launch(void* const* d_in, const int* in_sizes, int n_in,
                              void* d_out, int out_size, void* d_ws, size_t ws_size,
                              hipStream_t stream) {
  (void)in_sizes; (void)n_in; (void)out_size; (void)ws_size;
  const float* query  = (const float*)d_in[0];
  const float* key    = (const float*)d_in[1];
  const float* value  = (const float*)d_in[2];
  const float* pos    = (const float*)d_in[3];
  const unsigned char* mask = (const unsigned char*)d_in[4]; // bool [B,1,S]
  const float* Wq     = (const float*)d_in[5];
  const float* bq     = (const float*)d_in[6];
  const float* Wk     = (const float*)d_in[7];
  const float* bk     = (const float*)d_in[8];
  const float* Wv     = (const float*)d_in[9];
  const float* bv     = (const float*)d_in[10];
  const float* Wpos   = (const float*)d_in[11];
  const float* u_bias = (const float*)d_in[12];
  const float* v_bias = (const float*)d_in[13];
  const float* Wo     = (const float*)d_in[14];
  const float* bo     = (const float*)d_in[15];

  // Workspace layout (28 MiB): 5 bf16 operand copies (4 MiB each) + f32 ctx.
  char* ws = (char*)d_ws;
  const size_t MB = 1024u * 1024u;
  bf16*  qu  = (bf16*)(ws + 0 * MB);
  bf16*  qv  = (bf16*)(ws + 4 * MB);
  bf16*  kb  = (bf16*)(ws + 8 * MB);
  bf16*  vb_ = (bf16*)(ws + 12 * MB);
  bf16*  peb = (bf16*)(ws + 16 * MB);
  float* ctx = (float*)(ws + 20 * MB);

  float* out  = (float*)d_out;                     // [2,2048,512]
  float* attn = out + (size_t)M_TOTAL * D_MODEL;   // [2,16,2048,2048]

  // 1) Four projections (q with u/v biases baked into bf16 copies).
  proj_kernel<<<dim3(4, 256, 4), 256, 0, stream>>>(
      query, key, value, pos, Wq, Wk, Wv, Wpos, bq, bk, bv,
      u_bias, v_bias, qu, qv, kb, vb_, peb);

  // 2) Fused scores + rel-shift + mask + softmax. LDS = (17+16)*2048 floats.
  const size_t smem_bytes = (size_t)(17 + 16) * S_LEN * sizeof(float); // 264 KB
  score_softmax_kernel<<<dim3(S_LEN / 16, BS * N_HEADS), 256, smem_bytes, stream>>>(
      qu, qv, kb, peb, mask, attn);

  // 3) context = attn @ v (async-staged V panels in LDS).
  context_kernel<<<dim3(16, BS * N_HEADS), 256, 0, stream>>>(attn, vb_, ctx);

  // 4) out = ctx @ Wo^T + bo
  outproj_kernel<<<dim3(4, 256), 256, 0, stream>>>(ctx, Wo, bo, out);
}
